// Net_55662776156696
// MI455X (gfx1250) — compile-verified, matrix operand order checked by reading, not run on hardware
//
#include <hip/hip_runtime.h>
#include <hip/hip_bf16.h>

// ---------------------------------------------------------------------------
// GAT(8 heads x 8) -> MHI pooling -> GAT(1 head x 7) for gfx1250 (CDNA5).
// Big GEMM (100000x1433 @ 1433x64) runs on v_wmma_f32_16x16x32_bf16 with
// double-buffered async global->LDS staging of the W panel (ASYNCcnt path).
// ---------------------------------------------------------------------------

typedef __attribute__((ext_vector_type(16))) __bf16 v16bf;
typedef __attribute__((ext_vector_type(8)))  float  v8f;

#define IN_F 1433
#define NEG_SLOPE 0.2f
#define KC 192          // K-chunk staged in LDS (6 WMMA steps per chunk)
#define NCH 8           // ceil(1433/192)
#define WSTR 1536       // padded row stride of prepacked W panel (= NCH*KC)
#define LDS_STR 200     // padded LDS row stride (bf16): 16B-aligned, no conflicts

__device__ __forceinline__ unsigned short f2bf(float f) {
  return __builtin_bit_cast(unsigned short, static_cast<__bf16>(f));
}

// order-preserving float<->uint map for atomicMax-based segment_max
__device__ __forceinline__ unsigned enc_ord(float f) {
  unsigned u = __float_as_uint(f);
  return (u & 0x80000000u) ? ~u : (u | 0x80000000u);
}
__device__ __forceinline__ float dec_ord(unsigned u) {
  unsigned v = (u & 0x80000000u) ? (u & 0x7fffffffu) : ~u;
  return __uint_as_float(v);
}
__device__ __forceinline__ float lrelu(float x) {
  return x > 0.0f ? x : NEG_SLOPE * x;
}

// async 16B global->LDS copy, tracked by ASYNCcnt (gfx1250)
__device__ __forceinline__ void async_ld_b128(unsigned lds_off, const void* gptr) {
  asm volatile("global_load_async_to_lds_b128 %0, %1, off"
               :: "v"(lds_off), "v"(gptr) : "memory");
}
__device__ __forceinline__ void wait_asynccnt0() {
  asm volatile("s_wait_asynccnt 0x0" ::: "memory");
}

// ---------------------------------------------------------------------------
// K0: prepack W1[1433,64] -> wsW[64][WSTR] bf16, transposed + zero-padded
// ---------------------------------------------------------------------------
__global__ void wprep(const float* __restrict__ W, unsigned short* __restrict__ wsW)
{
  int i = blockIdx.x * blockDim.x + threadIdx.x;
  if (i >= 64 * WSTR) return;
  int n = i / WSTR, k = i % WSTR;
  float v = (k < IN_F) ? W[k * 64 + n] : 0.0f;
  wsW[i] = f2bf(v);
}

// ---------------------------------------------------------------------------
// K1: feat[N,64] = features[N,1433] @ W1[1433,64]  (bf16 WMMA, f32 accum)
// One wave = 16 rows x 64 cols (4 col-tiles share one A fragment per K-step).
// W panel chunks arrive in LDS via double-buffered async copies that overlap
// the WMMA stream; B frags are 2x ds_load_b128 in the native A/B layout.
// Main loop is branch-free (tail chunk isolated after the loop): row guard
// handled by clamping the source row, K tail by clamped-A x zero-padded-B.
// ---------------------------------------------------------------------------
__global__ void __launch_bounds__(256) gat1_gemm(
    const float* __restrict__ A, const unsigned short* __restrict__ Wbf,
    float* __restrict__ feat, int N)
{
  __shared__ unsigned short sB[2][64 * LDS_STR];   // 2 x 12800 bf16 = 51200 B

  const int lane = threadIdx.x & 31;
  const int wave = threadIdx.x >> 5;
  const int r    = lane & 15;
  const int hi   = lane >> 4;
  const int rowBase = (blockIdx.x * 8 + wave) * 16;
  const int row  = rowBase + r;
  const int rowC = row < N ? row : (N - 1);           // clamped load row
  const float* __restrict__ Arow = A + (long long)rowC * IN_F;
  const int kb   = hi * 8;            // lane-group K offset (A layout, 16-bit)

  // LDS byte offsets of the two buffers (flat shared addr low 32 bits)
  const unsigned lb0 = (unsigned)(unsigned long long)&sB[0][0];
  const unsigned lb1 = (unsigned)(unsigned long long)&sB[1][0];

  v8f c[4] = {};

  // issue async stage of chunk 0 into buffer 0
  for (int q = threadIdx.x; q < 64 * (KC / 8); q += 256) {  // 6 per thread
    int n  = q / (KC / 8);
    int qq = q % (KC / 8);
    async_ld_b128(lb0 + (unsigned)((n * LDS_STR + qq * 8) * 2),
                  Wbf + (size_t)n * WSTR + qq * 8);
  }

  // ---- full chunks 0..NCH-2: branch-free ---------------------------------
  for (int ch = 0; ch < NCH - 1; ++ch) {
    wait_asynccnt0();        // this wave's staged chunk has landed in LDS
    __syncthreads();         // everyone's chunk landed; prev compute finished

    // issue next chunk (always exists here) into the other buffer
    {
      const unsigned lbn = ((ch + 1) & 1) ? lb1 : lb0;
      const int nbase = (ch + 1) * KC;
      for (int q = threadIdx.x; q < 64 * (KC / 8); q += 256) {
        int n  = q / (KC / 8);
        int qq = q % (KC / 8);
        async_ld_b128(lbn + (unsigned)((n * LDS_STR + qq * 8) * 2),
                      Wbf + (size_t)n * WSTR + nbase + qq * 8);
      }
    }

    const unsigned short* sBc = &sB[ch & 1][0];
    const int base = ch * KC;

    for (int s = 0; s < KC / 32; ++s) {
      union { v16bf v; __bf16 h[16]; } ua;
      const float* ap = Arow + base + s * 32 + kb;
#pragma unroll
      for (int j = 0; j < 8; ++j) {
        ua.h[j]     = static_cast<__bf16>(ap[j]);
        ua.h[8 + j] = static_cast<__bf16>(ap[16 + j]);
      }
      const int klb = s * 32 + kb;
#pragma unroll
      for (int ct = 0; ct < 4; ++ct) {
        union { v16bf v; uint4 q[2]; } ub;
        const uint4* bp =
            reinterpret_cast<const uint4*>(sBc + (ct * 16 + r) * LDS_STR + klb);
        ub.q[0] = bp[0];  // halves 0..7  : K = klb .. klb+7
        ub.q[1] = bp[2];  // halves 8..15 : K = klb+16 .. klb+23
        c[ct] = __builtin_amdgcn_wmma_f32_16x16x32_bf16(
            false, ua.v, false, ub.v, (short)0, c[ct], false, false);
      }
    }
  }

  // ---- tail chunk: zero-padded B, address-clamped A ----------------------
  {
    wait_asynccnt0();
    __syncthreads();

    const unsigned short* sBc = &sB[(NCH - 1) & 1][0];
    const int base = (NCH - 1) * KC;
    const int nst  = (IN_F - base + 31) >> 5;         // 3

    for (int s = 0; s < nst; ++s) {
      union { v16bf v; __bf16 h[16]; } ua;
      const int ka0 = base + s * 32 + kb;
#pragma unroll
      for (int j = 0; j < 8; ++j) {
        int k1 = ka0 + j;       k1 = k1 < IN_F ? k1 : (IN_F - 1);
        int k2 = ka0 + 16 + j;  k2 = k2 < IN_F ? k2 : (IN_F - 1);
        ua.h[j]     = static_cast<__bf16>(Arow[k1]);  // B==0 there anyway
        ua.h[8 + j] = static_cast<__bf16>(Arow[k2]);
      }
      const int klb = s * 32 + kb;
#pragma unroll
      for (int ct = 0; ct < 4; ++ct) {
        union { v16bf v; uint4 q[2]; } ub;
        const uint4* bp =
            reinterpret_cast<const uint4*>(sBc + (ct * 16 + r) * LDS_STR + klb);
        ub.q[0] = bp[0];
        ub.q[1] = bp[2];
        c[ct] = __builtin_amdgcn_wmma_f32_16x16x32_bf16(
            false, ua.v, false, ub.v, (short)0, c[ct], false, false);
      }
    }
  }

  // C layout: VGPR i, lanes 0-15 -> M=i, lanes 16-31 -> M=8+i; N = lane&15
#pragma unroll
  for (int ct = 0; ct < 4; ++ct) {
    int col = ct * 16 + r;
#pragma unroll
    for (int i = 0; i < 8; ++i) {
      int rr = rowBase + hi * 8 + i;
      if (rr < N) feat[(long long)rr * 64 + col] = c[ct][i];
    }
  }
}

// ---------------------------------------------------------------------------
// K2: per (node, head): el/er projections + init emax/denom/rst
// ---------------------------------------------------------------------------
__global__ void prep1(const float* __restrict__ feat,
                      const float* __restrict__ attn_l1,
                      const float* __restrict__ attn_r1,
                      float* el, float* er, unsigned* emaxu,
                      float* denom, float* rst, int N)
{
  int i = blockIdx.x * blockDim.x + threadIdx.x;
  if (i >= N * 8) return;
  int n = i >> 3, h = i & 7;
  const float* f = feat + (long long)n * 64 + h * 8;
  float sl = 0.0f, sr = 0.0f;
#pragma unroll
  for (int o = 0; o < 8; ++o) {
    float v = f[o];
    sl += v * attn_l1[h * 8 + o];
    sr += v * attn_r1[h * 8 + o];
  }
  el[i] = sl;
  er[i] = sr;
  emaxu[i] = enc_ord(-__builtin_inff());
  denom[i] = 0.0f;
  float* rp = rst + (long long)n * 64 + h * 8;
#pragma unroll
  for (int o = 0; o < 8; ++o) rp[o] = 0.0f;
}

// ---------------------------------------------------------------------------
// K3/K4/K5: edge softmax + aggregation (8 heads)
// ---------------------------------------------------------------------------
__global__ void edge_max1(const int* __restrict__ src, const int* __restrict__ dst,
                          const float* __restrict__ el, const float* __restrict__ er,
                          unsigned* emaxu, int total)
{
  int i = blockIdx.x * blockDim.x + threadIdx.x;
  if (i >= total) return;
  int e = i >> 3, h = i & 7;
  int s = src[e], d = dst[e];
  float x = lrelu(el[s * 8 + h] + er[d * 8 + h]);
  atomicMax(&emaxu[d * 8 + h], enc_ord(x));
}

__global__ void edge_sum1(const int* __restrict__ src, const int* __restrict__ dst,
                          const float* __restrict__ el, const float* __restrict__ er,
                          const unsigned* __restrict__ emaxu, float* denom, int total)
{
  int i = blockIdx.x * blockDim.x + threadIdx.x;
  if (i >= total) return;
  int e = i >> 3, h = i & 7;
  int s = src[e], d = dst[e];
  float x = lrelu(el[s * 8 + h] + er[d * 8 + h]);
  float ex = __expf(x - dec_ord(emaxu[d * 8 + h]));
  atomicAdd(&denom[d * 8 + h], ex);
}

__global__ void edge_agg1(const int* __restrict__ src, const int* __restrict__ dst,
                          const float* __restrict__ el, const float* __restrict__ er,
                          const unsigned* __restrict__ emaxu,
                          const float* __restrict__ denom,
                          const float* __restrict__ feat, float* rst, int total)
{
  int i = blockIdx.x * blockDim.x + threadIdx.x;
  if (i >= total) return;
  int e = i >> 3, h = i & 7;
  int s = src[e], d = dst[e];
  float x = lrelu(el[s * 8 + h] + er[d * 8 + h]);
  float ex = __expf(x - dec_ord(emaxu[d * 8 + h]));
  float alpha = ex / denom[d * 8 + h];
  const float* fp = feat + (long long)s * 64 + h * 8;
  float* rp = rst + (long long)d * 64 + h * 8;
#pragma unroll
  for (int o = 0; o < 8; ++o) atomicAdd(&rp[o], fp[o] * alpha);
}

// ---------------------------------------------------------------------------
// K6: per node: relu(rst+b1) -> MHI attention pooling -> feat2/el2/er2
//     + init pass-2 softmax state and out = b2
// ---------------------------------------------------------------------------
__global__ void mhi_node(const float* __restrict__ rst, const float* __restrict__ b1,
                         const float* __restrict__ Wm, const float* __restrict__ bm,
                         const float* __restrict__ a,  const float* __restrict__ W2,
                         const float* __restrict__ attn_l2,
                         const float* __restrict__ attn_r2,
                         const float* __restrict__ b2,
                         float* feat2, float* el2, float* er2,
                         unsigned* emax2u, float* denom2, float* out, int N)
{
  int n = blockIdx.x * blockDim.x + threadIdx.x;
  if (n >= N) return;

  float x[8][8];
#pragma unroll
  for (int h = 0; h < 8; ++h)
#pragma unroll
    for (int o = 0; o < 8; ++o) {
      float v = rst[(long long)n * 64 + h * 8 + o] + b1[h * 8 + o];
      x[h][o] = v > 0.0f ? v : 0.0f;           // relu after GAT-1
    }

  // xl = x @ Wm^T + bm
  float xl[8][8];
#pragma unroll
  for (int h = 0; h < 8; ++h)
#pragma unroll
    for (int m = 0; m < 8; ++m) {
      float sum = bm[m];
#pragma unroll
      for (int j = 0; j < 8; ++j) sum += x[h][j] * Wm[m * 8 + j];
      xl[h][m] = sum;
    }

  float mean[8];
#pragma unroll
  for (int m = 0; m < 8; ++m) {
    float s = 0.0f;
#pragma unroll
    for (int h = 0; h < 8; ++h) s += xl[h][m];
    mean[m] = s * 0.125f;
  }
  float t2 = 0.0f;
#pragma unroll
  for (int m = 0; m < 8; ++m) t2 += mean[m] * a[8 + m];

  float ev[8], mx = -__builtin_inff();
#pragma unroll
  for (int h = 0; h < 8; ++h) {
    float s = t2;
#pragma unroll
    for (int m = 0; m < 8; ++m) s += xl[h][m] * a[m];
    s = s > 0.0f ? s : 0.0f;                   // relu
    ev[h] = s;
    mx = fmaxf(mx, s);
  }
  float ssum = 0.0f;
#pragma unroll
  for (int h = 0; h < 8; ++h) { ev[h] = __expf(ev[h] - mx); ssum += ev[h]; }
  float inv = 1.0f / ssum;

  float hv[8];
#pragma unroll
  for (int m = 0; m < 8; ++m) {
    float s = 0.0f;
#pragma unroll
    for (int h = 0; h < 8; ++h) s += x[h][m] * ev[h];
    hv[m] = s * inv;
  }

  // GAT-2 projection: feat2 = hv @ W2  (8 -> 7)
  float sl = 0.0f, sr = 0.0f;
#pragma unroll
  for (int o = 0; o < 7; ++o) {
    float s = 0.0f;
#pragma unroll
    for (int m = 0; m < 8; ++m) s += hv[m] * W2[m * 7 + o];
    feat2[(long long)n * 7 + o] = s;
    sl += s * attn_l2[o];
    sr += s * attn_r2[o];
    out[(long long)n * 7 + o] = b2[o];         // rst2 accumulates on top
  }
  el2[n] = sl;
  er2[n] = sr;
  emax2u[n] = enc_ord(-__builtin_inff());
  denom2[n] = 0.0f;
}

// ---------------------------------------------------------------------------
// K7/K8/K9: edge softmax + aggregation for GAT-2 (single head, 7 feats)
// ---------------------------------------------------------------------------
__global__ void edge_max2(const int* __restrict__ src, const int* __restrict__ dst,
                          const float* __restrict__ el2, const float* __restrict__ er2,
                          unsigned* emax2u, int E)
{
  int e = blockIdx.x * blockDim.x + threadIdx.x;
  if (e >= E) return;
  int s = src[e], d = dst[e];
  atomicMax(&emax2u[d], enc_ord(lrelu(el2[s] + er2[d])));
}

__global__ void edge_sum2(const int* __restrict__ src, const int* __restrict__ dst,
                          const float* __restrict__ el2, const float* __restrict__ er2,
                          const unsigned* __restrict__ emax2u, float* denom2, int E)
{
  int e = blockIdx.x * blockDim.x + threadIdx.x;
  if (e >= E) return;
  int s = src[e], d = dst[e];
  float x = lrelu(el2[s] + er2[d]);
  atomicAdd(&denom2[d], __expf(x - dec_ord(emax2u[d])));
}

__global__ void edge_agg2(const int* __restrict__ src, const int* __restrict__ dst,
                          const float* __restrict__ el2, const float* __restrict__ er2,
                          const unsigned* __restrict__ emax2u,
                          const float* __restrict__ denom2,
                          const float* __restrict__ feat2, float* out, int E)
{
  int e = blockIdx.x * blockDim.x + threadIdx.x;
  if (e >= E) return;
  int s = src[e], d = dst[e];
  float x = lrelu(el2[s] + er2[d]);
  float alpha = __expf(x - dec_ord(emax2u[d])) / denom2[d];
  const float* fp = feat2 + (long long)s * 7;
  float* op = out + (long long)d * 7;
#pragma unroll
  for (int o = 0; o < 7; ++o) atomicAdd(&op[o], fp[o] * alpha);
}

// ---------------------------------------------------------------------------
extern "C" void kernel_launch(void* const* d_in, const int* in_sizes, int n_in,
                              void* d_out, int out_size, void* d_ws, size_t ws_size,
                              hipStream_t stream)
{
  (void)n_in; (void)out_size; (void)ws_size;
  const float* features = (const float*)d_in[0];
  const int*   src      = (const int*)d_in[1];
  const int*   dst      = (const int*)d_in[2];
  const float* W1       = (const float*)d_in[3];
  const float* attn_l1  = (const float*)d_in[4];
  const float* attn_r1  = (const float*)d_in[5];
  const float* b1       = (const float*)d_in[6];
  const float* Wm       = (const float*)d_in[7];
  const float* bm       = (const float*)d_in[8];
  const float* a        = (const float*)d_in[9];
  const float* W2       = (const float*)d_in[10];
  const float* attn_l2  = (const float*)d_in[11];
  const float* attn_r2  = (const float*)d_in[12];
  const float* b2       = (const float*)d_in[13];
  float* out = (float*)d_out;

  const int N = in_sizes[0] / IN_F;   // 100000
  const int E = in_sizes[1];          // 1600000

  // workspace carve-up (~68 MB of f32 + 192 KB bf16 W panel)
  float* p = (float*)d_ws;
  float*    feat   = p;               p += (size_t)N * 64;
  float*    el     = p;               p += (size_t)N * 8;
  float*    er     = p;               p += (size_t)N * 8;
  unsigned* emaxu  = (unsigned*)p;    p += (size_t)N * 8;
  float*    denom  = p;               p += (size_t)N * 8;
  float*    rst    = p;               p += (size_t)N * 64;
  float*    feat2  = p;               p += (size_t)N * 7;
  float*    el2    = p;               p += N;
  float*    er2    = p;               p += N;
  unsigned* emax2u = (unsigned*)p;    p += N;
  float*    denom2 = p;               p += N;
  unsigned short* wsW = (unsigned short*)p;   // 64*WSTR bf16

  wprep<<<(64 * WSTR + 255) / 256, 256, 0, stream>>>(W1, wsW);

  const int tiles   = (N + 15) / 16;
  const int gblocks = (tiles + 7) / 8;     // 8 waves (16-row tiles) per block
  gat1_gemm<<<gblocks, 256, 0, stream>>>(features, wsW, feat, N);

  const int t1 = N * 8;
  prep1<<<(t1 + 255) / 256, 256, 0, stream>>>(feat, attn_l1, attn_r1,
                                              el, er, emaxu, denom, rst, N);

  const int t2 = E * 8;
  edge_max1<<<(t2 + 255) / 256, 256, 0, stream>>>(src, dst, el, er, emaxu, t2);
  edge_sum1<<<(t2 + 255) / 256, 256, 0, stream>>>(src, dst, el, er, emaxu, denom, t2);
  edge_agg1<<<(t2 + 255) / 256, 256, 0, stream>>>(src, dst, el, er, emaxu, denom,
                                                  feat, rst, t2);

  mhi_node<<<(N + 127) / 128, 128, 0, stream>>>(rst, b1, Wm, bm, a, W2,
                                                attn_l2, attn_r2, b2,
                                                feat2, el2, er2, emax2u, denom2,
                                                out, N);

  edge_max2<<<(E + 255) / 256, 256, 0, stream>>>(src, dst, el2, er2, emax2u, E);
  edge_sum2<<<(E + 255) / 256, 256, 0, stream>>>(src, dst, el2, er2, emax2u, denom2, E);
  edge_agg2<<<(E + 255) / 256, 256, 0, stream>>>(src, dst, el2, er2, emax2u, denom2,
                                                 feat2, out, E);
}
